// QuantizedSpectralConv2dV2_52896817217661
// MI455X (gfx1250) — matrix operand order repaired and checked by reading, not compiled, not run on hardware
//
#include <hip/hip_runtime.h>

typedef __attribute__((ext_vector_type(16))) _Float16 v16h;
typedef __attribute__((ext_vector_type(8)))  _Float16 v8h;
typedef __attribute__((ext_vector_type(8)))  float    v8f;

#define NB    16
#define NCIN  64
#define NCOUT 64
#define NH    256
#define NW    256
#define NKX   64            // 32 low + 32 high kx modes
#define NKY   32
#define NMODES (NKX*NKY)    // 2048
#define FROWS (NB*NCIN*NH)  // 262144
#define IROWS (NB*NCOUT*NH) // 262144

static __device__ __forceinline__ v8f wmma16(v16h a, v16h b, v8f c) {
  return __builtin_amdgcn_wmma_f32_16x16x32_f16(false, a, false, b, (short)0, c, false, false);
}

// ---- fragment helpers: K is innermost (contiguous) for A and B per lane ----
// A 16x32 f16, source row-major [M][K], ld = K-stride of M:
//   lane<16: M=lane, elems 0..7=K0..7, 8..15=K16..23 ; lane>=16: M=lane-16, K8..15/K24..31
static __device__ __forceinline__ v16h loadA_k(const _Float16* pa, int ldm, int lane) {
  int m = lane & 15;
  int kb = (lane >> 4) << 3;
  v8h a0 = *(const v8h*)(pa + (size_t)m * ldm + kb);
  v8h a1 = *(const v8h*)(pa + (size_t)m * ldm + kb + 16);
  v16h a;
#pragma unroll
  for (int j = 0; j < 8; ++j) { a[j] = a0[j]; a[8 + j] = a1[j]; }
  return a;
}
static __device__ __forceinline__ v16h loadA_k_f32(const float* pa, int ldm, int lane) {
  int m = lane & 15;
  int kb = (lane >> 4) << 3;
  v8f a0 = *(const v8f*)(pa + (size_t)m * ldm + kb);
  v8f a1 = *(const v8f*)(pa + (size_t)m * ldm + kb + 16);
  v16h a;
#pragma unroll
  for (int j = 0; j < 8; ++j) { a[j] = (_Float16)a0[j]; a[8 + j] = (_Float16)a1[j]; }
  return a;
}
// B 32x16 f16, source stored N-major [N][K] (i.e. B transposed), ldn = K-stride of N:
//   lane<16: N=lane, elems 0..7=K0..7, 8..15=K16..23 ; lane>=16: N=lane-16, K8..15/K24..31
static __device__ __forceinline__ v16h loadB_k(const _Float16* pb, int ldn, int lane) {
  int n = lane & 15;
  int kb = (lane >> 4) << 3;
  v8h b0 = *(const v8h*)(pb + (size_t)n * ldn + kb);
  v8h b1 = *(const v8h*)(pb + (size_t)n * ldn + kb + 16);
  v16h b;
#pragma unroll
  for (int j = 0; j < 8; ++j) { b[j] = b0[j]; b[8 + j] = b1[j]; }
  return b;
}
static __device__ __forceinline__ v16h hneg(v16h a) {
  v16h r;
#pragma unroll
  for (int j = 0; j < 16; ++j) r[j] = -a[j];
  return r;
}
// C/D 16x16 f32: VGPR r -> M=r+8*(lane>=16), N=lane&15
// store into [N][M] layout (ldm = M-stride of N): 8 contiguous halves per lane
static __device__ __forceinline__ void storeC_T_f16(_Float16* pc, int ldm, int lane, v8f c) {
  int n = lane & 15;
  int mb = (lane >> 4) << 3;
  v8h h;
#pragma unroll
  for (int r = 0; r < 8; ++r) h[r] = (_Float16)c[r];
  *(v8h*)(pc + (size_t)n * ldm + mb) = h;
}

// ---- twiddle tables (six 16384-elem f16 tables, 1/16 norm folded in) ----
// F1T [c=64][w=256] (B-transposed), T2 [kxp=64][h=256], T3 [h=256][kxp=64],
// T4T [w=256][j=64] (B^T of inverse-W DFT; hermitian doubling folded in)
__global__ __launch_bounds__(256) void k_init_twiddles(
    _Float16* F1T, _Float16* T2re, _Float16* T2im,
    _Float16* T3re, _Float16* T3im, _Float16* T4T) {
  int idx = blockIdx.x * 256 + threadIdx.x;
  int reg = idx >> 14;
  int t = idx & 16383;
  const float S = 1.0f / 16.0f;
  const float TP = 6.2831853071795864f;
  if (reg == 0) {            // F1T[c][w]: c<32 cos(2pi c w/256), c>=32 -sin
    int c = t >> 8, w = t & 255, k = c & 31;
    float th = TP * (float)((k * w) & 255) * (1.0f / 256.0f);
    F1T[t] = (_Float16)(((c < 32) ? __cosf(th) : -__sinf(th)) * S);
  } else if (reg <= 2) {     // T2[kxp][h]: e^{-2pi i kx h/256}
    int kxp = t >> 8, h = t & 255;
    int kx = (kxp < 32) ? kxp : (192 + kxp);
    float th = TP * (float)((kx * h) & 255) * (1.0f / 256.0f);
    if (reg == 1) T2re[t] = (_Float16)(__cosf(th) * S);
    else          T2im[t] = (_Float16)(-__sinf(th) * S);
  } else if (reg <= 4) {     // T3[h][kxp]: e^{+2pi i kx h/256}
    int h = t >> 6, kxp = t & 63;
    int kx = (kxp < 32) ? kxp : (192 + kxp);
    float th = TP * (float)((kx * h) & 255) * (1.0f / 256.0f);
    if (reg == 3) T3re[t] = (_Float16)(__cosf(th) * S);
    else          T3im[t] = (_Float16)(__sinf(th) * S);
  } else {                   // T4T[w][j]: j<32: c_k cos ; j>=32: -c_k sin
    int w = t >> 6, j = t & 63, k = j & 31;
    float ck = (k == 0) ? 1.0f : 2.0f;
    float th = TP * (float)((k * w) & 255) * (1.0f / 256.0f);
    T4T[t] = (_Float16)(((j < 32) ? ck * __cosf(th) : -ck * __sinf(th)) * S);
  }
}

// ---- dequant -> per-mode B-transposed planes wT[m][o][i] ----
__global__ __launch_bounds__(256) void k_dequant(
    const int* __restrict__ q1, const int* __restrict__ q2,
    const float* s1, const float* n1, const float* s2, const float* n2,
    _Float16* __restrict__ wreT, _Float16* __restrict__ wimT) {
  int idx = blockIdx.x * 256 + threadIdx.x;    // NMODES*64*64 = 8388608 exact
  int m = idx >> 12;
  int o = (idx >> 6) & 63;
  int i = idx & 63;
  int kx = m >> 5, ky = m & 31;
  const int* q; float sc, mn; int mx;
  if (kx < 32) { q = q1; sc = s1[0]; mn = n1[0]; mx = kx; }
  else         { q = q2; sc = s2[0]; mn = n2[0]; mx = kx - 32; }
  size_t src = ((((size_t)i * NCOUT + o) * 32 + mx) * 32 + ky) * 2;
  wreT[idx] = (_Float16)(((float)q[src]     + 127.0f) * sc + mn);
  wimT[idx] = (_Float16)(((float)q[src + 1] + 127.0f) * sc + mn);
}

// ---- stage 1: truncated DFT along W.  Y[ci][c][h], c<32 re(ky=c), c>=32 im ----
__global__ __launch_bounds__(256) void k_dft_w(
    const float* __restrict__ x, const _Float16* __restrict__ F1T,
    _Float16* __restrict__ Y) {
  int gw = (blockIdx.x * 256 + threadIdx.x) >> 5;      // 65536 waves
  int lane = threadIdx.x & 31;
  int mt = gw >> 2;                                    // 16384 row (h) tiles
  int nt = gw & 3;                                     // 4 col tiles
  int ci = mt >> 4;                                    // which (b,i) image
  int h0 = (mt & 15) << 4;
  const float* Ap = x + (size_t)mt * 16 * NW;
  const _Float16* Bp = F1T + (size_t)nt * 16 * NW;
  v8f acc = {};
#pragma unroll
  for (int ks = 0; ks < 8; ++ks) {                     // K = W = 256
    if (ks < 6) __builtin_prefetch(Ap + (ks + 2) * 32, 0, 1);
    v16h a = loadA_k_f32(Ap + ks * 32, NW, lane);
    v16h b = loadB_k(Bp + ks * 32, NW, lane);
    acc = wmma16(a, b, acc);
  }
  // C tile: M=h (contiguous per lane), N=c  -> Y[ci][c0+n][h0..] packed store
  storeC_T_f16(Y + (size_t)ci * 64 * NH + (size_t)(nt * 16) * NH + h0, NH, lane, acc);
}

// ---- stage 2: complex DFT along H onto kept kx; scatter to xft[m][b][i] ----
__global__ __launch_bounds__(256) void k_dft_h(
    const _Float16* __restrict__ Y,
    const _Float16* __restrict__ T2re, const _Float16* __restrict__ T2im,
    _Float16* __restrict__ xftre, _Float16* __restrict__ xftim) {
  int wv = threadIdx.x >> 5;                           // 8 waves; blockIdx = (b,i)
  int lane = threadIdx.x & 31;
  int ci = blockIdx.x;                                 // 0..1023 = b*64+i
  int bb = ci >> 6, ii = ci & 63;
  int mt = wv >> 1;                                    // kx tile 0..3
  int nt = wv & 1;                                     // ky tile 0..1
  const _Float16* Yb = Y + (size_t)ci * 64 * NH;       // [c][h]
  v8f accRe = {}, accIm = {};
#pragma unroll
  for (int ks = 0; ks < 8; ++ks) {                     // K = H = 256
    v16h ar = loadA_k(T2re + (size_t)mt * 16 * NH + ks * 32, NH, lane);
    v16h ai = loadA_k(T2im + (size_t)mt * 16 * NH + ks * 32, NH, lane);
    v16h br = loadB_k(Yb + (size_t)(nt * 16) * NH + ks * 32, NH, lane);
    v16h bi = loadB_k(Yb + (size_t)(32 + nt * 16) * NH + ks * 32, NH, lane);
    accRe = wmma16(ar, br, accRe);
    accRe = wmma16(hneg(ai), bi, accRe);
    accIm = wmma16(ar, bi, accIm);
    accIm = wmma16(ai, br, accIm);
  }
  int n = lane & 15, mb = (lane >> 4) << 3;
#pragma unroll
  for (int r = 0; r < 8; ++r) {
    int kx = mt * 16 + mb + r, ky = nt * 16 + n;
    size_t d = ((size_t)(kx * NKY + ky) * NB + bb) * NCIN + ii;
    xftre[d] = (_Float16)accRe[r];
    xftim[d] = (_Float16)accIm[r];
  }
}

// ---- stage 3: per-mode complex channel GEMM [16x64]x[64x64] -> oft[b][o][ky][kx] ----
__global__ __launch_bounds__(256) void k_modegemm(
    const _Float16* __restrict__ xftre, const _Float16* __restrict__ xftim,
    const _Float16* __restrict__ wreT, const _Float16* __restrict__ wimT,
    _Float16* __restrict__ oftre, _Float16* __restrict__ oftim) {
  int gw = (blockIdx.x * 256 + threadIdx.x) >> 5;      // 8192 waves
  int lane = threadIdx.x & 31;
  int m = gw >> 2, nt = gw & 3;
  int kx = m >> 5, ky = m & 31;
  const _Float16* Ar = xftre + (size_t)m * NB * NCIN;  // [b][i]
  const _Float16* Ai = xftim + (size_t)m * NB * NCIN;
  const _Float16* Br = wreT + (size_t)m * NCIN * NCOUT + (size_t)(nt * 16) * NCIN;
  const _Float16* Bi = wimT + (size_t)m * NCIN * NCOUT + (size_t)(nt * 16) * NCIN;
  v8f accRe = {}, accIm = {};
#pragma unroll
  for (int ks = 0; ks < 2; ++ks) {                     // K = Cin = 64
    v16h ar = loadA_k(Ar + ks * 32, NCIN, lane);
    v16h ai = loadA_k(Ai + ks * 32, NCIN, lane);
    v16h br = loadB_k(Br + ks * 32, NCIN, lane);
    v16h bi = loadB_k(Bi + ks * 32, NCIN, lane);
    accRe = wmma16(ar, br, accRe);
    accRe = wmma16(hneg(ai), bi, accRe);
    accIm = wmma16(ar, bi, accIm);
    accIm = wmma16(ai, br, accIm);
  }
  // C tile: rows b, cols o -> oft[(b*64+o)][ky][kx]
  int n = lane & 15, mb = (lane >> 4) << 3;
#pragma unroll
  for (int r = 0; r < 8; ++r) {
    size_t d = ((size_t)((mb + r) * NCOUT + nt * 16 + n) * NKY + ky) * NKX + kx;
    oftre[d] = (_Float16)accRe[r];
    oftim[d] = (_Float16)accIm[r];
  }
}

// ---- stage 4a: complex inverse DFT along H.  Z[row=(bo,h)][j], j<32 re, j>=32 im ----
__global__ __launch_bounds__(256) void k_idft_h(
    const _Float16* __restrict__ oftre, const _Float16* __restrict__ oftim,
    const _Float16* __restrict__ T3re, const _Float16* __restrict__ T3im,
    _Float16* __restrict__ Z) {
  int gw = (blockIdx.x * 256 + threadIdx.x) >> 5;      // 32768 waves
  int lane = threadIdx.x & 31;
  int bo = gw >> 5;                                    // 0..1023 = b*64+o
  int tl = gw & 31;
  int mt = tl >> 1, nt = tl & 1;                       // h tile, ky tile
  const _Float16* Fr = oftre + (size_t)bo * NKY * NKX; // [ky][kx]
  const _Float16* Fi = oftim + (size_t)bo * NKY * NKX;
  v8f accRe = {}, accIm = {};
#pragma unroll
  for (int ks = 0; ks < 2; ++ks) {                     // K = kx = 64
    v16h ar = loadA_k(T3re + (size_t)mt * 16 * NKX + ks * 32, NKX, lane);
    v16h ai = loadA_k(T3im + (size_t)mt * 16 * NKX + ks * 32, NKX, lane);
    v16h br = loadB_k(Fr + (size_t)(nt * 16) * NKX + ks * 32, NKX, lane);
    v16h bi = loadB_k(Fi + (size_t)(nt * 16) * NKX + ks * 32, NKX, lane);
    accRe = wmma16(ar, br, accRe);
    accRe = wmma16(hneg(ai), bi, accRe);
    accIm = wmma16(ar, bi, accIm);
    accIm = wmma16(ai, br, accIm);
  }
  int n = lane & 15, mb = (lane >> 4) << 3;
#pragma unroll
  for (int r = 0; r < 8; ++r) {
    size_t row = (size_t)bo * NH + mt * 16 + mb + r;
    Z[row * 64 + nt * 16 + n]      = (_Float16)accRe[r];
    Z[row * 64 + 32 + nt * 16 + n] = (_Float16)accIm[r];
  }
}

// ---- stage 4b (swapped orientation): out^T tile = T4T[w][j] x Z^T ----
// M=w(256), N=row(=b,o,h), K=j(64).  C store: 8 contiguous f32 per lane into d_out.
__global__ __launch_bounds__(256) void k_idft_w(
    const _Float16* __restrict__ Z, const _Float16* __restrict__ T4T,
    float* __restrict__ out) {
  int gw = (blockIdx.x * 256 + threadIdx.x) >> 5;      // 262144 waves
  int lane = threadIdx.x & 31;
  int ntr = gw >> 4;                                   // 16384 row tiles
  int mtw = gw & 15;                                   // 16 w tiles
  size_t row0 = (size_t)ntr * 16;
  const _Float16* Ap = T4T + (size_t)(mtw * 16) * 64;  // [w][j]
  const _Float16* Bp = Z + row0 * 64;                  // [row][j] (K innermost)
  __builtin_prefetch(Bp, 0, 1);
  v8f acc = {};
#pragma unroll
  for (int ks = 0; ks < 2; ++ks) {                     // K = 64
    v16h a = loadA_k(Ap + ks * 32, 64, lane);
    v16h b = loadB_k(Bp + ks * 32, 64, lane);
    acc = wmma16(a, b, acc);
  }
  int n = lane & 15, mb = (lane >> 4) << 3;
  *(v8f*)(out + (row0 + n) * NW + mtw * 16 + mb) = acc;
}

extern "C" void kernel_launch(void* const* d_in, const int* in_sizes, int n_in,
                              void* d_out, int out_size, void* d_ws, size_t ws_size,
                              hipStream_t stream) {
  (void)in_sizes; (void)n_in; (void)out_size; (void)ws_size;
  const float* x  = (const float*)d_in[0];
  const int*   q1 = (const int*)d_in[1];
  const int*   q2 = (const int*)d_in[2];
  const float* s1 = (const float*)d_in[3];
  const float* n1 = (const float*)d_in[4];
  const float* s2 = (const float*)d_in[5];
  const float* n2 = (const float*)d_in[6];
  float* out = (float*)d_out;

  _Float16* p = (_Float16*)d_ws;
  _Float16* F1T  = p; p += 16384;
  _Float16* T2re = p; p += 16384;
  _Float16* T2im = p; p += 16384;
  _Float16* T3re = p; p += 16384;
  _Float16* T3im = p; p += 16384;
  _Float16* T4T  = p; p += 16384;
  _Float16* Y    = p; p += (size_t)FROWS * 64;             // [ci][c][h]
  _Float16* xftre = p; p += (size_t)NMODES * NB * NCIN;    // [m][b][i]
  _Float16* xftim = p; p += (size_t)NMODES * NB * NCIN;
  _Float16* wreT  = p; p += (size_t)NMODES * NCIN * NCOUT; // [m][o][i]
  _Float16* wimT  = p; p += (size_t)NMODES * NCIN * NCOUT;
  _Float16* oftre = p; p += (size_t)NMODES * NB * NCOUT;   // [b][o][ky][kx]
  _Float16* oftim = p; p += (size_t)NMODES * NB * NCOUT;
  _Float16* Z     = p; p += (size_t)IROWS * 64;            // [row][j]

  k_init_twiddles<<<384,   256, 0, stream>>>(F1T, T2re, T2im, T3re, T3im, T4T);
  k_dequant      <<<32768, 256, 0, stream>>>(q1, q2, s1, n1, s2, n2, wreT, wimT);
  k_dft_w        <<<8192,  256, 0, stream>>>(x, F1T, Y);
  k_dft_h        <<<1024,  256, 0, stream>>>(Y, T2re, T2im, xftre, xftim);
  k_modegemm     <<<1024,  256, 0, stream>>>(xftre, xftim, wreT, wimT, oftre, oftim);
  k_idft_h       <<<4096,  256, 0, stream>>>(oftre, oftim, T3re, T3im, Z);
  k_idft_w       <<<32768, 256, 0, stream>>>(Z, T4T, out);
}
